// Ernie4Attention_66881230733992
// MI455X (gfx1250) — matrix operand order repaired and checked by reading, not compiled, not run on hardware
//
#include <hip/hip_runtime.h>
#include <hip/hip_bf16.h>

#define HIDDEN     4096
#define N_Q_HEADS  32
#define N_KV_HEADS 4
#define HEAD_DIM   128
#define Q_SIZE     4096
#define KV_SIZE    512
#define QKV_N      5120
#define BATCH      2
#define SEQ        2048
#define TOKENS     (BATCH * SEQ)
#define GQA_G      (N_Q_HEADS / N_KV_HEADS)

typedef __bf16 bf16_t;
typedef __attribute__((ext_vector_type(16))) __bf16 v16bf;
typedef __attribute__((ext_vector_type(8)))  __bf16 v8bf;
typedef __attribute__((ext_vector_type(4)))  __bf16 v4bf;
typedef __attribute__((ext_vector_type(8)))  float  v8f;

// ---------------- WMMA fragment helpers (wave32, 16x16x32 bf16) ----------------

// A-operand: 16(M) x 32(K). Lane l: row = l&15; lanes<16 hold K {0..7,16..23},
// lanes>=16 hold K {8..15,24..31}; element e -> VGPR e/2 half e%2.
__device__ __forceinline__ v16bf frag_a(const bf16_t* p, int ld) {
  const int lane = threadIdx.x & 31;
  const bf16_t* base = p + (size_t)(lane & 15) * ld + ((lane >> 4) << 3);
  v8bf lo = *(const v8bf*)(base);
  v8bf hi = *(const v8bf*)(base + 16);
  return __builtin_shufflevector(lo, hi, 0,1,2,3,4,5,6,7,8,9,10,11,12,13,14,15);
}

// B-operand: 32(K) x 16(N). Lane l: col n = l&15; lanes<16 hold K {0..7,16..23},
// lanes>=16 hold K {8..15,24..31}.
__device__ __forceinline__ v16bf frag_b(const bf16_t* p, int ld) {
  const int lane = threadIdx.x & 31;
  const bf16_t* base = p + (size_t)(lane & 15) * ld + ((lane >> 4) << 3);
  v8bf lo = *(const v8bf*)(base);
  v8bf hi = *(const v8bf*)(base + 16);
  return __builtin_shufflevector(lo, hi, 0,1,2,3,4,5,6,7,8,9,10,11,12,13,14,15);
}

__device__ __forceinline__ v8f wmma_bf16(v16bf a, v16bf b, v8f c) {
  return __builtin_amdgcn_wmma_f32_16x16x32_bf16(false, a, false, b, (short)0, c,
                                                 false, false);
}

// Async global->LDS 16-byte copy (gfx1250, tracked by ASYNCcnt).
__device__ __forceinline__ void async_copy_b128(unsigned lds_off,
                                                const void* gptr) {
  asm volatile("global_load_async_to_lds_b128 %0, %1, off"
               :: "v"(lds_off), "v"(gptr) : "memory");
}
#define WAIT_ASYNCCNT(n) asm volatile("s_wait_asynccnt " #n ::: "memory")

// ---------------- f32 -> bf16 conversion ----------------

__global__ void cvt_f32_to_bf16(const float* __restrict__ in,
                                bf16_t* __restrict__ out, int n4) {
  int i = blockIdx.x * blockDim.x + threadIdx.x;
  if (i >= n4) return;
  float4 f = ((const float4*)in)[i];
  v4bf o;
  o[0] = (bf16_t)f.x; o[1] = (bf16_t)f.y; o[2] = (bf16_t)f.z; o[3] = (bf16_t)f.w;
  ((v4bf*)out)[i] = o;
}

// ---------------- GEMM: C[M,N] = A[M,K] * Bm[N,K]^T (bf16 in, f32 acc) --------
// 256 threads = 8 waves; block tile 128(M) x 256(N); wave tile 64x64.
// Double-buffered async global->LDS staging; LDS rows padded to 40 elems (80B)
// so b128 fragment loads are bank-conflict-free (stride 20 banks mod 64).

#define LDA_PAD   40
#define A_STAGE   (128 * LDA_PAD)              // elements
#define B_STAGE   (256 * LDA_PAD)
#define STAGE_ELT (A_STAGE + B_STAGE)          // 15360 elements = 30720 B

template <bool OUT_F32>
__global__ __launch_bounds__(256, 1)
void gemm_wmma(const bf16_t* __restrict__ A, const bf16_t* __restrict__ Bm,
               void* __restrict__ Cout, int M, int N, int K) {
  __shared__ __align__(16) bf16_t smem[2 * STAGE_ELT];   // 60 KB

  const int tid  = threadIdx.x;
  const int wave = tid >> 5;
  const int lane = tid & 31;
  const int wm   = wave & 1;        // 2 waves along M
  const int wn   = wave >> 1;       // 4 waves along N
  const size_t mb = (size_t)blockIdx.x * 128;
  const size_t nb = (size_t)blockIdx.y * 256;

  const bf16_t* Abase = A  + mb * K;
  const bf16_t* Bbase = Bm + nb * K;

  // Stage one 128x32 A tile + 256x32 B tile into LDS buffer `buf`.
  // Per wave: exactly 6 async instructions (2 for A, 4 for B).
  auto issue_tile = [&](int kt, int buf) {
    const bf16_t* Ag = Abase + (size_t)kt * 32;
    const bf16_t* Bg = Bbase + (size_t)kt * 32;
    unsigned la = (unsigned)(uintptr_t)(smem + buf * STAGE_ELT);
    unsigned lb = la + A_STAGE * 2;             // bytes
#pragma unroll
    for (int i = 0; i < 2; ++i) {               // A: 512 chunks of 16B
      int c = tid * 2 + i, row = c >> 2, ch = c & 3;
      async_copy_b128(la + (unsigned)(row * LDA_PAD + ch * 8) * 2,
                      Ag + (size_t)row * K + ch * 8);
    }
#pragma unroll
    for (int i = 0; i < 4; ++i) {               // B: 1024 chunks of 16B
      int c = tid * 4 + i, row = c >> 2, ch = c & 3;
      async_copy_b128(lb + (unsigned)(row * LDA_PAD + ch * 8) * 2,
                      Bg + (size_t)row * K + ch * 8);
    }
  };

  const v8f zero = {0.f, 0.f, 0.f, 0.f, 0.f, 0.f, 0.f, 0.f};
  v8f acc[4][4];
#pragma unroll
  for (int i = 0; i < 4; ++i)
#pragma unroll
    for (int j = 0; j < 4; ++j) acc[i][j] = zero;

  const int KT = K / 32;
  issue_tile(0, 0);

  for (int kt = 0; kt < KT; ++kt) {
    const int buf = kt & 1;
    if (kt + 1 < KT) {
      issue_tile(kt + 1, buf ^ 1);
      WAIT_ASYNCCNT(6);       // own writes of tile kt complete (next 6 in flight)
    } else {
      WAIT_ASYNCCNT(0);
    }
    __syncthreads();          // publish LDS writes across waves

    const bf16_t* sA = smem + buf * STAGE_ELT;
    const bf16_t* sB = sA + A_STAGE;
    v16bf af[4], bfv[4];
#pragma unroll
    for (int i = 0; i < 4; ++i)
      af[i] = frag_a(sA + (wm * 64 + 16 * i) * LDA_PAD, LDA_PAD);
#pragma unroll
    for (int j = 0; j < 4; ++j)
      bfv[j] = frag_b(sB + (wn * 64 + 16 * j) * LDA_PAD, LDA_PAD);
#pragma unroll
    for (int i = 0; i < 4; ++i)
#pragma unroll
      for (int j = 0; j < 4; ++j) acc[i][j] = wmma_bf16(af[i], bfv[j], acc[i][j]);

    __syncthreads();          // done reading `buf` before it is re-staged
  }

  const int col = lane & 15;
  const int rh  = (lane >> 4) << 3;
#pragma unroll
  for (int i = 0; i < 4; ++i)
#pragma unroll
    for (int j = 0; j < 4; ++j)
#pragma unroll
      for (int r = 0; r < 8; ++r) {
        size_t row = mb + wm * 64 + 16 * i + rh + r;
        size_t cc  = nb + wn * 64 + 16 * j + col;
        if constexpr (OUT_F32)
          ((float*)Cout)[row * N + cc] = acc[i][j][r];
        else
          ((bf16_t*)Cout)[row * N + cc] = (bf16_t)acc[i][j][r];
      }
}

// ---------------- RoPE (interleaved / GPT-J) + head-major repack --------------

__global__ void rope_pack_q(const bf16_t* __restrict__ qkv,
                            const int* __restrict__ pos,
                            bf16_t* __restrict__ Qo) {
  int idx = blockIdx.x * blockDim.x + threadIdx.x;
  const int total = TOKENS * N_Q_HEADS * (HEAD_DIM / 2);
  if (idx >= total) return;
  int pr = idx % (HEAD_DIM / 2);
  int h  = (idx / (HEAD_DIM / 2)) % N_Q_HEADS;
  int t  = idx / ((HEAD_DIM / 2) * N_Q_HEADS);
  int b = t / SEQ, s = t % SEQ;
  float p = (float)pos[s];
  float freq = __powf(500000.0f, -((float)(2 * pr)) * (1.0f / 128.0f));
  float sn, cs;
  __sincosf(p * freq, &sn, &cs);
  const bf16_t* src = qkv + (size_t)t * QKV_N + h * HEAD_DIM + 2 * pr;
  float e = (float)src[0], od = (float)src[1];
  bf16_t* dst = Qo + (((size_t)(b * N_Q_HEADS + h)) * SEQ + s) * HEAD_DIM + 2 * pr;
  dst[0] = (bf16_t)(e * cs - od * sn);
  dst[1] = (bf16_t)(od * cs + e * sn);
}

__global__ void rope_pack_k(const bf16_t* __restrict__ qkv,
                            const int* __restrict__ pos,
                            bf16_t* __restrict__ Ko) {
  int idx = blockIdx.x * blockDim.x + threadIdx.x;
  const int total = TOKENS * N_KV_HEADS * (HEAD_DIM / 2);
  if (idx >= total) return;
  int pr = idx % (HEAD_DIM / 2);
  int h  = (idx / (HEAD_DIM / 2)) % N_KV_HEADS;
  int t  = idx / ((HEAD_DIM / 2) * N_KV_HEADS);
  int b = t / SEQ, s = t % SEQ;
  float p = (float)pos[s];
  float freq = __powf(500000.0f, -((float)(2 * pr)) * (1.0f / 128.0f));
  float sn, cs;
  __sincosf(p * freq, &sn, &cs);
  const bf16_t* src = qkv + (size_t)t * QKV_N + Q_SIZE + h * HEAD_DIM + 2 * pr;
  float e = (float)src[0], od = (float)src[1];
  bf16_t* dst = Ko + (((size_t)(b * N_KV_HEADS + h)) * SEQ + s) * HEAD_DIM + 2 * pr;
  dst[0] = (bf16_t)(e * cs - od * sn);
  dst[1] = (bf16_t)(od * cs + e * sn);
}

// V -> transposed layout [B, Hkv, D, S] so PV B-operand loads are contiguous.
__global__ void pack_vt(const bf16_t* __restrict__ qkv, bf16_t* __restrict__ Vt) {
  int idx = blockIdx.x * blockDim.x + threadIdx.x;
  const int total = TOKENS * KV_SIZE;
  if (idx >= total) return;
  int c = idx % KV_SIZE;
  int t = idx / KV_SIZE;
  int b = t / SEQ, s = t % SEQ;
  int hv = c / HEAD_DIM, d = c % HEAD_DIM;
  Vt[(((size_t)(b * N_KV_HEADS + hv)) * HEAD_DIM + d) * SEQ + s] =
      qkv[(size_t)t * QKV_N + Q_SIZE + KV_SIZE + c];
}

// ---------------- Flash attention (causal GQA), WMMA + online softmax ---------
// 128 threads = 4 waves; each wave owns 16 query rows of one (b, head).

__global__ __launch_bounds__(128, 1)
void attn_wmma(const bf16_t* __restrict__ Q,   // [B, Hq, S, D]
               const bf16_t* __restrict__ Km,  // [B, Hkv, S, D]
               const bf16_t* __restrict__ Vt,  // [B, Hkv, D, S]
               bf16_t* __restrict__ ctx) {     // [B, S, Hq*D]
  __shared__ __align__(16) bf16_t Pbuf[4 * 16 * 40];  // per-wave 16x32 (ld=40)

  const int wave = threadIdx.x >> 5;
  const int lane = threadIdx.x & 31;
  const int qblk = blockIdx.x % (SEQ / 64);
  const int h    = (blockIdx.x / (SEQ / 64)) % N_Q_HEADS;
  const int b    = blockIdx.x / ((SEQ / 64) * N_Q_HEADS);
  const int q0   = qblk * 64 + wave * 16;
  const int hk   = h / GQA_G;

  const bf16_t* Qb = Q  + ((size_t)(b * N_Q_HEADS + h) * SEQ + q0) * HEAD_DIM;
  const bf16_t* Kb = Km + (size_t)(b * N_KV_HEADS + hk) * SEQ * HEAD_DIM;
  const bf16_t* Vb = Vt + (size_t)(b * N_KV_HEADS + hk) * HEAD_DIM * SEQ;
  bf16_t* Pw = Pbuf + wave * 16 * 40;

  v16bf qf[4];
#pragma unroll
  for (int i = 0; i < 4; ++i) qf[i] = frag_a(Qb + 32 * i, HEAD_DIM);

  const v8f zero = {0.f, 0.f, 0.f, 0.f, 0.f, 0.f, 0.f, 0.f};
  v8f o[8];
#pragma unroll
  for (int t = 0; t < 8; ++t) o[t] = zero;
  float m_i[8], l_i[8];
#pragma unroll
  for (int r = 0; r < 8; ++r) { m_i[r] = -3.0e38f; l_i[r] = 0.f; }

  const float scale = 0.08838834764831845f;  // 1/sqrt(128)
  const int col = lane & 15;
  const int rh  = (lane >> 4) << 3;

  for (int kt = 0; kt < q0 + 16; kt += 32) {
    // S-tiles: 16 q-rows x 32 keys
    v8f s0 = zero, s1 = zero;
#pragma unroll
    for (int i = 0; i < 4; ++i) {
      v16bf k0f = frag_b(Kb + (size_t)kt * HEAD_DIM + 32 * i, HEAD_DIM);
      v16bf k1f = frag_b(Kb + (size_t)(kt + 16) * HEAD_DIM + 32 * i, HEAD_DIM);
      s0 = wmma_bf16(qf[i], k0f, s0);
      s1 = wmma_bf16(qf[i], k1f, s1);
    }
    // causal mask + online softmax; row lives across 16 lanes of a half
#pragma unroll
    for (int r = 0; r < 8; ++r) {
      int qrow = q0 + rh + r;
      float v0 = (kt + col      <= qrow) ? s0[r] * scale : -3.0e38f;
      float v1 = (kt + 16 + col <= qrow) ? s1[r] * scale : -3.0e38f;
      float mx = fmaxf(v0, v1);
#pragma unroll
      for (int off = 1; off < 16; off <<= 1)
        mx = fmaxf(mx, __shfl_xor(mx, off, 32));
      float mnew = fmaxf(m_i[r], mx);
      float corr = __expf(m_i[r] - mnew);
      float p0 = __expf(v0 - mnew);
      float p1 = __expf(v1 - mnew);
      float rs = p0 + p1;
#pragma unroll
      for (int off = 1; off < 16; off <<= 1) rs += __shfl_xor(rs, off, 32);
      l_i[r] = l_i[r] * corr + rs;
      m_i[r] = mnew;
#pragma unroll
      for (int t = 0; t < 8; ++t) o[t][r] = o[t][r] * corr;
      Pw[(rh + r) * 40 + col]      = (bf16_t)p0;
      Pw[(rh + r) * 40 + 16 + col] = (bf16_t)p1;
    }
    asm volatile("s_wait_dscnt 0" ::: "memory");  // DS RAW before frag_a reload
    v16bf pf = frag_a(Pw, 40);
#pragma unroll
    for (int t = 0; t < 8; ++t) {
      v16bf vf = frag_b(Vb + (size_t)(t * 16) * SEQ + kt, SEQ);
      o[t] = wmma_bf16(pf, vf, o[t]);
    }
  }

#pragma unroll
  for (int t = 0; t < 8; ++t)
#pragma unroll
    for (int r = 0; r < 8; ++r) {
      int qrow = q0 + rh + r;
      ctx[((size_t)(b * SEQ + qrow)) * Q_SIZE + h * HEAD_DIM + t * 16 + col] =
          (bf16_t)(o[t][r] / l_i[r]);
    }
}

// ---------------- launcher ----------------

extern "C" void kernel_launch(void* const* d_in, const int* in_sizes, int n_in,
                              void* d_out, int out_size, void* d_ws, size_t ws_size,
                              hipStream_t stream) {
  const int*   pos    = (const int*)d_in[0];
  const float* hidden = (const float*)d_in[1];
  const float* w_qkv  = (const float*)d_in[2];
  const float* w_o    = (const float*)d_in[3];

  char* ws = (char*)d_ws;
  const size_t SZ_WQKV = (size_t)QKV_N  * HIDDEN * 2;   // 40 MB
  const size_t SZ_HID  = (size_t)TOKENS * HIDDEN * 2;   // 32 MB
  const size_t SZ_QKV  = (size_t)TOKENS * QKV_N  * 2;   // 40 MB
  const size_t SZ_WO   = (size_t)HIDDEN * Q_SIZE * 2;   // 32 MB
  const size_t SZ_K    = (size_t)BATCH * N_KV_HEADS * SEQ * HEAD_DIM * 2;  // 4 MB

  bf16_t* wq_bf  = (bf16_t*)(ws);
  bf16_t* hid_bf = (bf16_t*)(ws + SZ_WQKV);
  bf16_t* qkv_bf = (bf16_t*)(ws + SZ_WQKV + SZ_HID);
  bf16_t* wo_bf  = (bf16_t*)(ws + SZ_WQKV + SZ_HID + SZ_QKV);
  bf16_t* K_bf   = (bf16_t*)(ws + SZ_WQKV + SZ_HID + SZ_QKV + SZ_WO);
  bf16_t* Vt_bf  = (bf16_t*)(ws + SZ_WQKV + SZ_HID + SZ_QKV + SZ_WO + SZ_K);
  bf16_t* Q_bf   = hid_bf;  // reuse: hidden dead after QKV GEMM
  bf16_t* ctx_bf = wq_bf;   // reuse: w_qkv dead after QKV GEMM

  const int th = 256;
  auto nb = [](long long n, int t) { return (unsigned)((n + t - 1) / t); };

  cvt_f32_to_bf16<<<nb((long long)TOKENS * HIDDEN / 4, th), th, 0, stream>>>(
      hidden, hid_bf, TOKENS * HIDDEN / 4);
  cvt_f32_to_bf16<<<nb((long long)QKV_N * HIDDEN / 4, th), th, 0, stream>>>(
      w_qkv, wq_bf, QKV_N * HIDDEN / 4);
  cvt_f32_to_bf16<<<nb((long long)HIDDEN * Q_SIZE / 4, th), th, 0, stream>>>(
      w_o, wo_bf, HIDDEN * Q_SIZE / 4);

  dim3 g1(TOKENS / 128, QKV_N / 256);
  gemm_wmma<false><<<g1, 256, 0, stream>>>(hid_bf, wq_bf, qkv_bf,
                                           TOKENS, QKV_N, HIDDEN);

  rope_pack_q<<<nb((long long)TOKENS * N_Q_HEADS * (HEAD_DIM / 2), th), th, 0,
                stream>>>(qkv_bf, pos, Q_bf);
  rope_pack_k<<<nb((long long)TOKENS * N_KV_HEADS * (HEAD_DIM / 2), th), th, 0,
                stream>>>(qkv_bf, pos, K_bf);
  pack_vt<<<nb((long long)TOKENS * KV_SIZE, th), th, 0, stream>>>(qkv_bf, Vt_bf);

  attn_wmma<<<BATCH * N_Q_HEADS * (SEQ / 64), 128, 0, stream>>>(Q_bf, K_bf,
                                                                Vt_bf, ctx_bf);

  dim3 g2(TOKENS / 128, HIDDEN / 256);
  gemm_wmma<true><<<g2, 256, 0, stream>>>(ctx_bf, wo_bf, d_out,
                                          TOKENS, HIDDEN, Q_SIZE);
}
